// Loss_87454124081368
// MI455X (gfx1250) — matrix operand order repaired and checked by reading, not compiled, not run on hardware
//
#include <hip/hip_runtime.h>
#include <hip/hip_bf16.h>

typedef __attribute__((ext_vector_type(16))) __bf16    v16bf;
typedef __attribute__((ext_vector_type(8)))  unsigned  v8u;
typedef __attribute__((ext_vector_type(8)))  float     v8f;

#define KC      128   // K elements staged per block iteration
#define BSTRIDE 136   // padded bf16 LDS row stride (shorts): 272B/row -> bank start +4/row
#define KBLK    512   // split-K chunk per block (multiple of KC; all HW are multiples of 512)

// fp32 -> bf16 round-to-nearest-even
__device__ __forceinline__ unsigned short f2bf(float f) {
    unsigned u = __builtin_bit_cast(unsigned, f);
    u += 0x7FFFu + ((u >> 16) & 1u);
    return (unsigned short)(u >> 16);
}

__device__ __forceinline__ void wait_async0() {
#if __has_builtin(__builtin_amdgcn_s_wait_asynccnt)
    __builtin_amdgcn_s_wait_asynccnt(0);
#else
    asm volatile("s_wait_asynccnt 0x0" ::: "memory");
#endif
}

__device__ __forceinline__ void block_reduce_atomic(float v, float scale, float* out) {
    __shared__ float red[256];
    red[threadIdx.x] = v;
    __syncthreads();
    for (int s = 128; s > 0; s >>= 1) {
        if ((int)threadIdx.x < s) red[threadIdx.x] += red[threadIdx.x + s];
        __syncthreads();
    }
    if (threadIdx.x == 0) atomicAdd(out, red[0] * scale);
}

// ---------------------------------------------------------------------------
// fp32 -> bf16 slab conversion: ybf[C][slabW] = bf16(x[C][HW][k0 : k0+slabW])
// ---------------------------------------------------------------------------
__global__ __launch_bounds__(256) void cvt_bf16_kernel(
    const float* __restrict__ x, int HW, int k0, int slabW, int C,
    unsigned short* __restrict__ ybf)
{
    int w4 = slabW >> 2;
    int total4 = C * w4;
    for (int i = blockIdx.x * 256 + threadIdx.x; i < total4;
         i += gridDim.x * 256) {
        int row = i / w4;
        int c4  = (i - row * w4) << 2;
        float4 v = *(const float4*)(x + (size_t)row * HW + k0 + c4);
        unsigned short* o = ybf + (size_t)row * slabW + c4;
        o[0] = f2bf(v.x);  o[1] = f2bf(v.y);
        o[2] = f2bf(v.z);  o[3] = f2bf(v.w);
    }
}

// ---------------------------------------------------------------------------
// Signed Gram accumulation on a bf16 panel: gout += sign * (X @ X^T) for one
// 64x64 output block (4x4 WMMA tiles, 8 waves x 2 tiles), split-K on blockIdx.y.
// xbf is [C, kWidth] row-major bf16 (stride ldK).
// Staging: GLOBAL_LOAD_ASYNC_TO_LDS_B128 -> 128 rows x 128 K bf16 in LDS.
// Compute: V_WMMA_F32_16X16X32_BF16, fragments are pure ds_load_b128 (no cvt).
// ---------------------------------------------------------------------------
__global__ __launch_bounds__(256) void gram_bf16_wmma_kernel(
    const unsigned short* __restrict__ xbf, int C, int ldK, int kWidth,
    float sign, float* __restrict__ gout)
{
    __shared__ unsigned short sT[128 * BSTRIDE];  // rows 0-63: A panel, 64-127: B panel

    const int nP = C >> 6;                 // 64-wide panels per dimension
    const int ti = blockIdx.x / nP;
    const int tj = blockIdx.x % nP;
    const int kStart = blockIdx.y * KBLK;
    const int kEnd   = min(kWidth, kStart + KBLK);

    const int t       = threadIdx.x;
    const int w       = t >> 5;            // wave id 0..7
    const int lane    = t & 31;
    const int halfSel = lane >> 4;
    const int mn      = lane & 15;
    const int mi      = w >> 1;            // wave's tile row 0..3
    const int nb      = (w & 1) * 2;       // wave's tile col pair {0,1} or {2,3}

    const unsigned short* aBase = xbf + (size_t)(ti * 64) * ldK;
    const unsigned short* bBase = xbf + (size_t)(tj * 64) * ldK;
    const unsigned ldsBase = (unsigned)(uintptr_t)&sT[0];

    v8f acc0 = {}, acc1 = {};

    for (int kc = kStart; kc < kEnd; kc += KC) {
        // ---- async DMA staging: 128 rows x 128 K bf16 (32KB), 8 b128/thread ----
#pragma unroll
        for (int it = 0; it < 8; ++it) {
            int idx  = t + it * 256;       // 0..2047 b128 slots
            int row  = idx >> 4;           // 16 slots per 128-short row
            int col8 = (idx & 15) << 3;    // short offset 0..120
            const unsigned short* src =
                (row < 64 ? aBase + (size_t)row * ldK
                          : bBase + (size_t)(row - 64) * ldK) + kc + col8;
            unsigned lds = ldsBase + (unsigned)(row * BSTRIDE + col8) * 2u;
            asm volatile("global_load_async_to_lds_b128 %0, %1, off"
                         :: "v"(lds), "v"((unsigned long long)(uintptr_t)src)
                         : "memory");
        }
        wait_async0();
        __syncthreads();

        const unsigned short* aRow  = &sT[(mi * 16 + mn) * BSTRIDE];
        const unsigned short* bRow0 = &sT[(64 + nb * 16 + mn) * BSTRIDE];
        const unsigned short* bRow1 = bRow0 + 16 * BSTRIDE;

#pragma unroll
        for (int ks = 0; ks < KC / 32; ++ks) {
            const int kOff = ks * 32;
            // A 16x32 bf16: lanes 0-15 K=0..15, lanes 16-31 K=16..31
            uint4 alo = *(const uint4*)(aRow + kOff + halfSel * 16);
            uint4 ahi = *(const uint4*)(aRow + kOff + halfSel * 16 + 8);
            // B 32x16 bf16: lanes 0-15 K=0..15, lanes 16-31 K=16..31
            uint4 b0lo = *(const uint4*)(bRow0 + kOff + halfSel * 16);
            uint4 b0hi = *(const uint4*)(bRow0 + kOff + halfSel * 16 + 8);
            uint4 b1lo = *(const uint4*)(bRow1 + kOff + halfSel * 16);
            uint4 b1hi = *(const uint4*)(bRow1 + kOff + halfSel * 16 + 8);

            v8u au, b0u, b1u;
            au[0] = alo.x;  au[1] = alo.y;  au[2] = alo.z;  au[3] = alo.w;
            au[4] = ahi.x;  au[5] = ahi.y;  au[6] = ahi.z;  au[7] = ahi.w;
            b0u[0] = b0lo.x; b0u[1] = b0lo.y; b0u[2] = b0lo.z; b0u[3] = b0lo.w;
            b0u[4] = b0hi.x; b0u[5] = b0hi.y; b0u[6] = b0hi.z; b0u[7] = b0hi.w;
            b1u[0] = b1lo.x; b1u[1] = b1lo.y; b1u[2] = b1lo.z; b1u[3] = b1lo.w;
            b1u[4] = b1hi.x; b1u[5] = b1hi.y; b1u[6] = b1hi.z; b1u[7] = b1hi.w;

            v16bf a  = __builtin_bit_cast(v16bf, au);
            v16bf b0 = __builtin_bit_cast(v16bf, b0u);
            v16bf b1 = __builtin_bit_cast(v16bf, b1u);
            acc0 = __builtin_amdgcn_wmma_f32_16x16x32_bf16(
                false, a, false, b0, (short)0, acc0, false, false);
            acc1 = __builtin_amdgcn_wmma_f32_16x16x32_bf16(
                false, a, false, b1, (short)0, acc1, false, false);
        }
        __syncthreads();   // all fragment reads done before next stage overwrites LDS
    }

    // ---- epilogue: each wave owns its two 16x16 tiles -> direct atomics ----
    {
        const int m0   = ti * 64 + mi * 16 + 8 * halfSel;  // + r below
        const int col0 = tj * 64 + nb * 16 + mn;
#pragma unroll
        for (int r = 0; r < 8; ++r) {
            atomicAdd(&gout[(size_t)(m0 + r) * C + col0],      sign * acc0[r]);
            atomicAdd(&gout[(size_t)(m0 + r) * C + col0 + 16], sign * acc1[r]);
        }
    }
}

// ---------------------------------------------------------------------------
__global__ __launch_bounds__(256) void zero_kernel(float* ws, int n, float* out) {
    for (int i = blockIdx.x * blockDim.x + threadIdx.x; i < n;
         i += gridDim.x * blockDim.x)
        ws[i] = 0.f;
    if (blockIdx.x == 0 && threadIdx.x == 0) out[0] = 0.f;
}

__global__ __launch_bounds__(256) void sqdiff_kernel(
    const float* __restrict__ a, const float* __restrict__ b, int n,
    float scale, float* out)
{
    float acc = 0.f;
    for (int i = blockIdx.x * blockDim.x + threadIdx.x; i < n;
         i += gridDim.x * blockDim.x) {
        float d = a[i] - b[i];
        acc += d * d;
    }
    block_reduce_atomic(acc, scale, out);
}

__global__ __launch_bounds__(256) void sqsum_kernel(
    const float* __restrict__ g, int n, float scale, float* out)
{
    float acc = 0.f;
    for (int i = blockIdx.x * blockDim.x + threadIdx.x; i < n;
         i += gridDim.x * blockDim.x) {
        float d = g[i];
        acc += d * d;
    }
    block_reduce_atomic(acc, scale, out);
}

// ---------------------------------------------------------------------------
extern "C" void kernel_launch(void* const* d_in, const int* in_sizes, int n_in,
                              void* d_out, int out_size, void* d_ws, size_t ws_size,
                              hipStream_t stream)
{
    (void)in_sizes; (void)n_in; (void)out_size;

    const float* S[5];
    const float* P[5];
    for (int i = 0; i < 5; ++i) {
        S[i] = (const float*)d_in[i];
        P[i] = (const float*)d_in[5 + i];
    }
    const float* p5 = (const float*)d_in[10];
    const float* c0 = (const float*)d_in[11];
    float* outp  = (float*)d_out;
    float* gdiff = (float*)d_ws;

    const int Cs[5]  = {64, 128, 256, 512, 512};
    const int HWs[5] = {512 * 512, 256 * 256, 128 * 128, 64 * 64, 32 * 32};

    size_t goff[5];
    size_t tot = 0;
    for (int i = 0; i < 5; ++i) { goff[i] = tot; tot += (size_t)Cs[i] * Cs[i]; }

    // bf16 slab scratch lives after the (256B-aligned) Gram-diff buffers
    size_t gramPad = ((tot * sizeof(float)) + 255) & ~(size_t)255;
    unsigned short* bfbuf = (unsigned short*)((char*)d_ws + gramPad);
    size_t availShorts = (ws_size > gramPad) ? (ws_size - gramPad) / 2 : 0;

    // 1) zero Gram-diff workspace and the output scalar
    zero_kernel<<<dim3(512), dim3(256), 0, stream>>>(gdiff, (int)tot, outp);

    // 2) per scale: slab-convert to bf16, then signed Gram accumulation
    for (int i = 0; i < 5; ++i) {
        int C = Cs[i], HW = HWs[i];
        int nP = C / 64;

        long maxW = (C > 0) ? (long)(availShorts / (size_t)C) : 0;
        int slabW = (maxW < (long)HW) ? (int)maxW : HW;
        slabW = (slabW / KBLK) * KBLK;            // multiple of KBLK
        if (slabW < KBLK) slabW = KBLK;           // minimum slab (~C*1KB of ws)
        if (slabW > HW) slabW = HW;

        for (int k0 = 0; k0 < HW; k0 += slabW) {
            int wdt = (HW - k0 < slabW) ? (HW - k0) : slabW;  // multiple of KBLK
            dim3 ggrid(nP * nP, wdt / KBLK);
            int cvtBlocks = (C * (wdt / 4) + 255) / 256;
            if (cvtBlocks > 1024) cvtBlocks = 1024;

            cvt_bf16_kernel<<<dim3(cvtBlocks), dim3(256), 0, stream>>>(
                P[i], HW, k0, wdt, C, bfbuf);
            gram_bf16_wmma_kernel<<<ggrid, dim3(256), 0, stream>>>(
                bfbuf, C, wdt, wdt, +1.0f, gdiff + goff[i]);

            cvt_bf16_kernel<<<dim3(cvtBlocks), dim3(256), 0, stream>>>(
                S[i], HW, k0, wdt, C, bfbuf);
            gram_bf16_wmma_kernel<<<ggrid, dim3(256), 0, stream>>>(
                bfbuf, C, wdt, wdt, -1.0f, gdiff + goff[i]);
        }
    }

    // 3) content loss: ALPHA * 0.5 * sum((c0 - p5)^2)
    {
        int n = 512 * 64 * 64;
        sqdiff_kernel<<<dim3(1024), dim3(256), 0, stream>>>(c0, p5, n, 0.5f, outp);
    }

    // 4) style loss: BETA * w_i / (4 C^2 (HW)^2) * sum(diff^2)
    for (int i = 0; i < 5; ++i) {
        double C = (double)Cs[i], HW = (double)HWs[i];
        float scale = (float)(1000.0 * 0.2 / (4.0 * C * C * HW * HW));
        int n = Cs[i] * Cs[i];
        int blocks = (n + 255) / 256;
        if (blocks > 1024) blocks = 1024;
        sqsum_kernel<<<dim3(blocks), dim3(256), 0, stream>>>(gdiff + goff[i], n,
                                                             scale, outp);
    }
}
